// DecoderWithAttention_8426725834794
// MI455X (gfx1250) — compile-verified, matrix operand order checked by reading, not run on hardware
//
#include <hip/hip_runtime.h>

// ---------------------------------------------------------------------------
// Problem constants (from reference)
// ---------------------------------------------------------------------------
#define BSZ   64
#define PIX   196          // 14*14
#define ENC   2048
#define DD    512          // DEC_D == ATT_D == EMB_D
#define VOC   10000
#define TSTEPS 51          // MAX_LEN-1
#define CAPL  52
#define XW    3072         // packed x row: [emb(512) | awe(2048) | h(512)]

typedef unsigned short u16;                                   // bf16 storage
typedef __attribute__((ext_vector_type(16))) __bf16 bf16x16;  // WMMA A/B frag
typedef __attribute__((ext_vector_type(8)))  float  f32x8;    // WMMA C/D frag

__device__ __forceinline__ u16 f2bf(float x) {
    union { float f; unsigned int u; } v; v.f = x;
    unsigned int u = v.u;
    u += 0x7fffu + ((u >> 16) & 1u);      // round-to-nearest-even
    return (u16)(u >> 16);
}
__device__ __forceinline__ float bf2f(u16 h) {
    union { float f; unsigned int u; } v; v.u = ((unsigned int)h) << 16;
    return v.f;
}
__device__ __forceinline__ float sigf(float x) { return 1.0f / (1.0f + __expf(-x)); }

// ---------------------------------------------------------------------------
// bf16 WMMA GEMM:  C(M,N) = A(M,K)bf16 @ B(K,N)bf16 + bias[N]
//   A row-major (stride lda); B column-major (Bt[n*ldb + k]).
//   Each wave computes a 32 x (NJ*16) block: 2 M-tiles x NJ N-tiles.
//   NJ is a template constant so all accumulator loops fully unroll ->
//   accumulators stay in SSA registers (no v_movrel indexed-VGPR traffic).
//   All control flow is wave-uniform so EXEC is all-ones at every v_wmma.
//   M must be a multiple of 32; N a multiple of 16 (true for all calls).
// ---------------------------------------------------------------------------
union FragU { bf16x16 v; uint4 q[2]; };

template <int NJ>
__global__ void gemm_bf16(const u16* __restrict__ A, int lda,
                          const u16* __restrict__ Bt, int ldb,
                          const float* __restrict__ bias,
                          const float* __restrict__ rowScale,
                          float* __restrict__ C, int ldc,
                          int M, int K,
                          int gcount,     // #n-groups in this launch
                          int tnBase)     // first n-tile of this launch
{
    const int lane = threadIdx.x & 31;
    const int wave = threadIdx.x >> 5;
    const int wpb  = blockDim.x >> 5;
    const int mt2  = M >> 5;                       // 32-row m-groups
    const int group = blockIdx.x * wpb + wave;
    if (group >= mt2 * gcount) return;             // wave-uniform
    const int gm  = group / gcount;
    const int tn0 = tnBase + (group % gcount) * NJ;

    const int khalf = (lane & 16) ? 8 : 0;         // per-lane K sub-block
    const u16* aRow0 = A + (size_t)(gm * 32 + (lane & 15)) * lda + khalf;
    const u16* aRow1 = aRow0 + (size_t)16 * lda;

    f32x8 acc0[NJ] = {};
    f32x8 acc1[NJ] = {};
    for (int k0 = 0; k0 < K; k0 += 32) {
        FragU a0, a1;
        a0.q[0] = *(const uint4*)(const void*)(aRow0 + k0);
        a0.q[1] = *(const uint4*)(const void*)(aRow0 + k0 + 16);
        a1.q[0] = *(const uint4*)(const void*)(aRow1 + k0);
        a1.q[1] = *(const uint4*)(const void*)(aRow1 + k0 + 16);
        #pragma unroll
        for (int j = 0; j < NJ; ++j) {
            const int col = (tn0 + j) * 16 + (lane & 15);
            const u16* bCol = Bt + (size_t)col * ldb + khalf + k0;
            FragU b;
            b.q[0] = *(const uint4*)(const void*)(bCol);
            b.q[1] = *(const uint4*)(const void*)(bCol + 16);
            acc0[j] = __builtin_amdgcn_wmma_f32_16x16x32_bf16(
                          false, a0.v, false, b.v, (short)0, acc0[j], false, false);
            acc1[j] = __builtin_amdgcn_wmma_f32_16x16x32_bf16(
                          false, a1.v, false, b.v, (short)0, acc1[j], false, false);
        }
    }

    // C/D layout: VGPR r -> row = r + 8*(lane>=16), col = lane%16
    const int rb0 = gm * 32 + ((lane & 16) ? 8 : 0);
    const int cin = lane & 15;
    #pragma unroll
    for (int j = 0; j < NJ; ++j) {
        const int colg = (tn0 + j) * 16 + cin;
        const float bv = bias ? bias[colg] : 0.0f;
        #pragma unroll
        for (int r = 0; r < 8; ++r) {
            int rg = rb0 + r;
            float v = acc0[j][r] + bv;
            if (rowScale) v *= rowScale[rg];
            C[(size_t)rg * ldc + colg] = v;
        }
        #pragma unroll
        for (int r = 0; r < 8; ++r) {
            int rg = rb0 + 16 + r;
            float v = acc1[j][r] + bv;
            if (rowScale) v *= rowScale[rg];
            C[(size_t)rg * ldc + colg] = v;
        }
    }
}

// ---------------------------------------------------------------------------
// Setup / utility kernels
// ---------------------------------------------------------------------------
__global__ void conv_bf16(const float* __restrict__ src, u16* __restrict__ dst,
                          long n, int cols, int dstStride)
{
    long i = (long)blockIdx.x * blockDim.x + threadIdx.x;
    if (i >= n) return;
    long r = i / cols; int c = (int)(i % cols);
    dst[r * (long)dstStride + c] = f2bf(src[i]);
}

// W (K x N, row-major f32) -> Wt (N x K, bf16)  [column-major B for GEMM]
__global__ void transpose_bf16(const float* __restrict__ W, u16* __restrict__ Wt,
                               int K, int N)
{
    long i = (long)blockIdx.x * blockDim.x + threadIdx.x;
    if (i >= (long)K * N) return;
    int n = (int)(i / K), k = (int)(i % K);
    Wt[i] = f2bf(W[(size_t)k * N + n]);
}

// Fused [W_ih (2560x2048); W_hh (512x2048)] -> Wt (2048 x 3072, bf16 col-major)
__global__ void build_wcat(const float* __restrict__ Wih, const float* __restrict__ Whh,
                           u16* __restrict__ Wt)
{
    long i = (long)blockIdx.x * blockDim.x + threadIdx.x;
    if (i >= 2048L * XW) return;
    int n = (int)(i / XW), k = (int)(i % XW);
    float v = (k < 2560) ? Wih[(size_t)k * 2048 + n] : Whh[(size_t)(k - 2560) * 2048 + n];
    Wt[i] = f2bf(v);
}

__global__ void mean_kernel(const float* __restrict__ enc,
                            float* __restrict__ mean, u16* __restrict__ meanB)
{
    int b = blockIdx.x;
    int d = blockIdx.y * blockDim.x + threadIdx.x;   // 0..2047
    const float* p = enc + (size_t)b * PIX * ENC + d;
    float s = 0.0f;
    for (int i = 0; i < PIX; ++i) s += p[(size_t)i * ENC];
    float m = s * (1.0f / (float)PIX);
    mean[b * ENC + d]  = m;
    meanB[b * ENC + d] = f2bf(m);
}

__global__ void add_vec(const float* a, const float* b, float* o, int n)
{
    int i = blockIdx.x * blockDim.x + threadIdx.x;
    if (i < n) o[i] = a[i] + b[i];
}

__global__ void int_to_float(const int* in, float* out, int n, int sub)
{
    int i = blockIdx.x * blockDim.x + threadIdx.x;
    if (i < n) out[i] = (float)(in[i] - sub);
}

__global__ void emb_gather(const int* __restrict__ caps, const float* __restrict__ emb,
                           u16* __restrict__ xcat, int t)
{
    int idx = blockIdx.x * blockDim.x + threadIdx.x;   // 64*512
    int b = idx >> 9, k = idx & 511;
    int tok = caps[b * CAPL + t];
    xcat[(size_t)b * XW + k] = f2bf(emb[(size_t)tok * DD + k]);
}

// ---------------------------------------------------------------------------
// Attention: e = tanh(att_enc + att_dec) @ W_full + b; alpha = softmax_p(e)
// one block per batch (8 waves, ~25 pixels each; wave reduces K=512)
// ---------------------------------------------------------------------------
__global__ void attention_kernel(const float* __restrict__ att_enc,   // (B,P,512)
                                 const float* __restrict__ att_dec,   // (B,512)
                                 const float* __restrict__ w_full,    // (512)
                                 const float* __restrict__ b_full,    // (1)
                                 const int*   __restrict__ lens,
                                 float* __restrict__ alpha,           // (B,P)
                                 float* __restrict__ alphas_out,      // (B,T,P)
                                 int t)
{
    __shared__ float e_s[PIX];
    __shared__ float m_s, sum_s;
    const int b    = blockIdx.x;
    const int tid  = threadIdx.x;          // 256
    const int lane = tid & 31, wave = tid >> 5;

    const float* ae = att_enc + (size_t)b * PIX * DD;
    const float* ad = att_dec + (size_t)b * DD;
    float wl[16], adl[16];
    #pragma unroll
    for (int i = 0; i < 16; ++i) { wl[i] = w_full[lane * 16 + i]; adl[i] = ad[lane * 16 + i]; }

    for (int p = wave; p < PIX; p += 8) {
        const float* row = ae + (size_t)p * DD;
        float s = 0.0f;
        #pragma unroll
        for (int i = 0; i < 16; ++i) s += tanhf(row[lane * 16 + i] + adl[i]) * wl[i];
        #pragma unroll
        for (int off = 16; off > 0; off >>= 1) s += __shfl_xor(s, off, 32);
        if (lane == 0) e_s[p] = s + b_full[0];
    }
    __syncthreads();
    if (tid == 0) {
        float m = -1e30f;
        for (int p = 0; p < PIX; ++p) m = fmaxf(m, e_s[p]);
        float su = 0.0f;
        for (int p = 0; p < PIX; ++p) su += __expf(e_s[p] - m);
        m_s = m; sum_s = su;
    }
    __syncthreads();
    const float mm = (t < lens[b] - 1) ? 1.0f : 0.0f;
    for (int p = tid; p < PIX; p += blockDim.x) {
        float a = __expf(e_s[p] - m_s) / sum_s;
        alpha[b * PIX + p] = a;
        alphas_out[((size_t)b * TSTEPS + t) * PIX + p] = a * mm;
    }
}

// awe[b,d] = sigmoid(gateRaw[b,d]) * sum_p alpha[b,p]*enc[b,p,d] -> x slot (bf16)
__global__ void awe_kernel(const float* __restrict__ alpha, const u16* __restrict__ encB,
                           const float* __restrict__ gateRaw, u16* __restrict__ xcat)
{
    const int b = blockIdx.x;
    const int d = blockIdx.y * blockDim.x + threadIdx.x;   // 0..2047
    const u16*  e  = encB + (size_t)b * PIX * ENC + d;
    const float* al = alpha + b * PIX;
    float acc = 0.0f;
    for (int p = 0; p < PIX; ++p) acc += al[p] * bf2f(e[(size_t)p * ENC]);
    float g = sigf(gateRaw[(size_t)b * ENC + d]);
    xcat[(size_t)b * XW + DD + d] = f2bf(g * acc);
}

// LSTM pointwise: gates (B,2048) -> h,c update with step mask; also packs h bf16
__global__ void lstm_kernel(const float* __restrict__ gates,
                            float* __restrict__ h, float* __restrict__ c,
                            u16* __restrict__ xcat, float* __restrict__ maskBuf,
                            const int* __restrict__ lens, int t)
{
    const int b = blockIdx.x;
    const int k = blockIdx.y * blockDim.x + threadIdx.x;   // 0..511
    const float* g4 = gates + (size_t)b * 2048;
    float ig = sigf(g4[k]);
    float fg = sigf(g4[512 + k]);
    float gg = tanhf(g4[1024 + k]);
    float og = sigf(g4[1536 + k]);
    float cn = fg * c[b * DD + k] + ig * gg;
    float hn = og * tanhf(cn);
    bool  m  = t < lens[b] - 1;
    float hv = m ? hn : h[b * DD + k];
    float cv = m ? cn : c[b * DD + k];
    h[b * DD + k] = hv;
    c[b * DD + k] = cv;
    xcat[(size_t)b * XW + 2560 + k] = f2bf(hv);
    if (k == 0) maskBuf[b] = m ? 1.0f : 0.0f;
}

// ---------------------------------------------------------------------------
// Host side
// ---------------------------------------------------------------------------
static inline void gemm(hipStream_t s, const u16* A, int lda, const u16* Bt, int ldb,
                        const float* bias, const float* rowScale, float* C, int ldc,
                        int M, int N, int K)
{
    const int mt2  = M / 32;           // M is a multiple of 32 in all calls
    const int nt   = N / 16;
    const int full = nt / 4;           // n-groups of 4 tiles
    const int rem  = nt - full * 4;    // 0..3 tail tiles
    if (full) {
        int G = mt2 * full;
        gemm_bf16<4><<<(G + 3) / 4, 128, 0, s>>>(A, lda, Bt, ldb, bias, rowScale,
                                                 C, ldc, M, K, full, 0);
    }
    if (rem) {
        int G = mt2;                   // gcount == 1
        dim3 blk((G + 3) / 4);
        switch (rem) {
        case 1: gemm_bf16<1><<<blk, 128, 0, s>>>(A, lda, Bt, ldb, bias, rowScale,
                                                 C, ldc, M, K, 1, full * 4); break;
        case 2: gemm_bf16<2><<<blk, 128, 0, s>>>(A, lda, Bt, ldb, bias, rowScale,
                                                 C, ldc, M, K, 1, full * 4); break;
        default: gemm_bf16<3><<<blk, 128, 0, s>>>(A, lda, Bt, ldb, bias, rowScale,
                                                  C, ldc, M, K, 1, full * 4); break;
        }
    }
}

extern "C" void kernel_launch(void* const* d_in, const int* in_sizes, int n_in,
                              void* d_out, int out_size, void* d_ws, size_t ws_size,
                              hipStream_t stream)
{
    // ---- inputs (setup_inputs order) ----
    const float* enc       = (const float*)d_in[0];
    const int*   caps      = (const int*)d_in[1];
    const int*   lens      = (const int*)d_in[2];
    const float* emb       = (const float*)d_in[3];
    const float* W_enc_att = (const float*)d_in[4];
    const float* b_enc_att = (const float*)d_in[5];
    const float* W_dec_att = (const float*)d_in[6];
    const float* b_dec_att = (const float*)d_in[7];
    const float* W_full    = (const float*)d_in[8];
    const float* b_full    = (const float*)d_in[9];
    const float* W_init_h  = (const float*)d_in[10];
    const float* b_init_h  = (const float*)d_in[11];
    const float* W_init_c  = (const float*)d_in[12];
    const float* b_init_c  = (const float*)d_in[13];
    const float* W_fbeta   = (const float*)d_in[14];
    const float* b_fbeta   = (const float*)d_in[15];
    const float* W_ih      = (const float*)d_in[16];
    const float* W_hh      = (const float*)d_in[17];
    const float* b_ih      = (const float*)d_in[18];
    const float* b_hh      = (const float*)d_in[19];
    const float* W_fc      = (const float*)d_in[20];
    const float* b_fc      = (const float*)d_in[21];

    // ---- outputs (tuple flat, return order) ----
    float* out_pred = (float*)d_out;                             // B*T*VOC
    float* out_caps = out_pred + (size_t)BSZ * TSTEPS * VOC;     // B*CAPL
    float* out_len  = out_caps + BSZ * CAPL;                     // B
    float* out_alph = out_len + BSZ;                             // B*T*PIX

    // ---- workspace carve-out (256B aligned) ----
    char* cur = (char*)d_ws;
    auto alloc = [&](size_t bytes) -> void* {
        void* p = cur; cur += (bytes + 255) & ~(size_t)255; return p;
    };
    u16*   encB    = (u16*)  alloc((size_t)BSZ * PIX * ENC * 2);
    float* att_enc = (float*)alloc((size_t)BSZ * PIX * DD * 4);
    u16*   WencT   = (u16*)  alloc((size_t)ENC * DD * 2);
    u16*   WdecT   = (u16*)  alloc((size_t)DD * DD * 2);
    u16*   WinitHT = (u16*)  alloc((size_t)ENC * DD * 2);
    u16*   WinitCT = (u16*)  alloc((size_t)ENC * DD * 2);
    u16*   WfbetaT = (u16*)  alloc((size_t)DD * ENC * 2);
    u16*   WcatT   = (u16*)  alloc((size_t)2048 * XW * 2);
    u16*   WfcT    = (u16*)  alloc((size_t)DD * VOC * 2);
    float* meanF   = (float*)alloc((size_t)BSZ * ENC * 4);
    u16*   meanB   = (u16*)  alloc((size_t)BSZ * ENC * 2);
    float* hbuf    = (float*)alloc((size_t)BSZ * DD * 4);
    float* cbuf    = (float*)alloc((size_t)BSZ * DD * 4);
    u16*   xcat    = (u16*)  alloc((size_t)BSZ * XW * 2);
    float* attdec  = (float*)alloc((size_t)BSZ * DD * 4);
    float* alpha   = (float*)alloc((size_t)BSZ * PIX * 4);
    float* gateRaw = (float*)alloc((size_t)BSZ * ENC * 4);
    float* gatesB  = (float*)alloc((size_t)BSZ * 2048 * 4);
    float* bcomb   = (float*)alloc((size_t)2048 * 4);
    float* maskBuf = (float*)alloc((size_t)BSZ * 4);
    (void)ws_size; (void)in_sizes; (void)n_in; (void)out_size;

    // ---- one-time setup ----
    {   // enc -> bf16
        long n = (long)BSZ * PIX * ENC;
        conv_bf16<<<(int)((n + 255) / 256), 256, 0, stream>>>(enc, encB, n, ENC, ENC);
    }
    auto tconv = [&](const float* W, u16* Wt, int K, int N) {
        long n = (long)K * N;
        transpose_bf16<<<(int)((n + 255) / 256), 256, 0, stream>>>(W, Wt, K, N);
    };
    tconv(W_enc_att, WencT, ENC, DD);
    tconv(W_dec_att, WdecT, DD, DD);
    tconv(W_init_h,  WinitHT, ENC, DD);
    tconv(W_init_c,  WinitCT, ENC, DD);
    tconv(W_fbeta,   WfbetaT, DD, ENC);
    tconv(W_fc,      WfcT,    DD, VOC);
    build_wcat<<<(int)((2048L * XW + 255) / 256), 256, 0, stream>>>(W_ih, W_hh, WcatT);
    add_vec<<<8, 256, 0, stream>>>(b_ih, b_hh, bcomb, 2048);
    int_to_float<<<(BSZ * CAPL + 255) / 256, 256, 0, stream>>>(caps, out_caps, BSZ * CAPL, 0);
    int_to_float<<<1, 64, 0, stream>>>(lens, out_len, BSZ, 1);

    mean_kernel<<<dim3(BSZ, ENC / 256), 256, 0, stream>>>(enc, meanF, meanB);
    // h0 / c0:  mean(64x2048) @ W_init(2048x512)
    gemm(stream, meanB, ENC, WinitHT, ENC, b_init_h, nullptr, hbuf, DD, BSZ, DD, ENC);
    gemm(stream, meanB, ENC, WinitCT, ENC, b_init_c, nullptr, cbuf, DD, BSZ, DD, ENC);
    // pack h0 into x's h slot (bf16, stride XW)
    conv_bf16<<<(BSZ * DD + 255) / 256, 256, 0, stream>>>(hbuf, xcat + 2560,
                                                          (long)BSZ * DD, DD, XW);
    // att_enc = enc @ W_enc_att + b   (12544 x 512, K=2048) — the big GEMM
    gemm(stream, encB, ENC, WencT, ENC, b_enc_att, nullptr, att_enc, DD,
         BSZ * PIX, DD, ENC);

    // ---- sequential decode loop ----
    for (int t = 0; t < TSTEPS; ++t) {
        // att_dec = h @ W_dec_att + b          (64x512, K=512)
        gemm(stream, xcat + 2560, XW, WdecT, DD, b_dec_att, nullptr,
             attdec, DD, BSZ, DD, DD);
        // gateRaw = h @ W_fbeta + b            (64x2048, K=512)
        gemm(stream, xcat + 2560, XW, WfbetaT, DD, b_fbeta, nullptr,
             gateRaw, ENC, BSZ, ENC, DD);
        // attention scores + softmax (+ masked alphas out)
        attention_kernel<<<BSZ, 256, 0, stream>>>(att_enc, attdec, W_full, b_full,
                                                  lens, alpha, out_alph, t);
        // emb_t into x slot
        emb_gather<<<(BSZ * DD + 255) / 256, 256, 0, stream>>>(caps, emb, xcat, t);
        // gated attention-weighted encoding into x slot
        awe_kernel<<<dim3(BSZ, ENC / 256), 256, 0, stream>>>(alpha, encB, gateRaw, xcat);
        // gates = [emb|awe|h] @ Wcat + (b_ih+b_hh)   (64x2048, K=3072)
        gemm(stream, xcat, XW, WcatT, XW, bcomb, nullptr, gatesB, 2048,
             BSZ, 2048, XW);
        // LSTM pointwise update (h, c, packed bf16 h, step mask)
        lstm_kernel<<<dim3(BSZ, 2), 256, 0, stream>>>(gatesB, hbuf, cbuf, xcat,
                                                      maskBuf, lens, t);
        // preds = (h @ W_fc + b_fc) * mask     (64x10000, K=512) -> out
        gemm(stream, xcat + 2560, XW, WfcT, DD, b_fc, maskBuf,
             out_pred + (size_t)t * VOC, TSTEPS * VOC, BSZ, VOC, DD);
    }
}
